// RGCNKG_85237920956626
// MI455X (gfx1250) — compile-verified, matrix operand order checked by reading, not run on hardware
//
#include <hip/hip_runtime.h>

// ---------------- problem constants (from reference) ----------------
#define N_ENT   10000
#define NUM_R   24        // R2 = num_relations * 2
#define RP      25        // +1 slot for the root ("self") matrix
#define N_BASES 30
#define EDGES   800000
#define D_IN    200       // embedding_dim
#define D_HID   100       // hidden_dim
// padded dims for WMMA (K multiple of 32, out multiple of 16)
#define K1P     224       // pad(200, 32)
#define O1P     112       // pad(100, 16)
#define K2P     128       // pad(100, 32)
#define O2P     208       // pad(200, 16)

typedef __bf16 bf16_t;
typedef __attribute__((ext_vector_type(16))) __bf16 v16bf;
typedef __attribute__((ext_vector_type(8)))  __bf16 v8bf;
typedef __attribute__((ext_vector_type(8)))  float  v8f;

// ---------------- small utility kernels ----------------
__global__ void k_zero_f32(float* __restrict__ p, int n) {
    int i = blockIdx.x * blockDim.x + threadIdx.x;
    if (i < n) p[i] = 0.0f;
}

// per-(dst, rel) in-degree counting
__global__ void k_count(const int* __restrict__ dst, const int* __restrict__ et,
                        float* __restrict__ cnt) {
    int e = blockIdx.x * blockDim.x + threadIdx.x;
    if (e < EDGES) atomicAdd(&cnt[dst[e] * NUM_R + et[e]], 1.0f);
}

__global__ void k_inv(const float* __restrict__ cnt, float* __restrict__ inv, int n) {
    int i = blockIdx.x * blockDim.x + threadIdx.x;
    if (i < n) inv[i] = 1.0f / fmaxf(cnt[i], 1.0f);
}

// Build transposed, zero-padded bf16 weights:
//   wt[r][o][k] = sum_b comp[r,b] * bases[b,k,o]   (r < NUM_R)
//   wt[NUM_R][o][k] = root[k,o]
// Layout [RP][op][kp] so WMMA B-fragments are contiguous along k.
__global__ void k_build_wt(const float* __restrict__ bases, const float* __restrict__ comp,
                           const float* __restrict__ root, bf16_t* __restrict__ wt,
                           int din, int dout, int kp, int op) {
    int idx = blockIdx.x * blockDim.x + threadIdx.x;
    int total = RP * op * kp;
    if (idx >= total) return;
    int k = idx % kp;
    int rem = idx / kp;
    int o = rem % op;
    int r = rem / op;
    float v = 0.0f;
    if (k < din && o < dout) {
        if (r < NUM_R) {
            #pragma unroll 5
            for (int b = 0; b < N_BASES; ++b)
                v += comp[r * N_BASES + b] * bases[(b * din + k) * dout + o];
        } else {
            v = root[k * dout + o];
        }
    }
    wt[idx] = (bf16_t)v;
}

// f32 -> bf16 with K zero-padding
__global__ void k_cvt_bf16(const float* __restrict__ x, bf16_t* __restrict__ xb,
                           int din, int kp) {
    int idx = blockIdx.x * blockDim.x + threadIdx.x;
    if (idx >= N_ENT * kp) return;
    int k = idx % kp;
    int n = idx / kp;
    xb[idx] = (k < din) ? (bf16_t)x[n * din + k] : (bf16_t)0.0f;
}

// relu(h) -> bf16 with K zero-padding (between layers)
__global__ void k_relu_cvt(const float* __restrict__ h, bf16_t* __restrict__ hb) {
    int idx = blockIdx.x * blockDim.x + threadIdx.x;
    if (idx >= N_ENT * K2P) return;
    int k = idx % K2P;
    int n = idx / K2P;
    float v = (k < D_HID) ? fmaxf(h[n * D_HID + k], 0.0f) : 0.0f;
    hb[idx] = (bf16_t)v;
}

// out[n, o] = xw[n, NUM_R(root slot), o] + bias[o]
__global__ void k_init_root(const float* __restrict__ xw, const float* __restrict__ bias,
                            float* __restrict__ out, int op, int dout) {
    int idx = blockIdx.x * blockDim.x + threadIdx.x;
    if (idx >= N_ENT * dout) return;
    int o = idx % dout;
    int n = idx / dout;
    out[idx] = xw[(n * RP + NUM_R) * op + o] + bias[o];
}

// ---------------- WMMA GEMM: xw[n, r, :] = x[n, :] @ Wt[r]^T ----------------
// One wave computes one 16x16 f32 tile via v_wmma_f32_16x16x32_bf16.
// A fragment (ISA 7.12.2, 16-bit A 16x32): lane<16 holds row=lane, K in
// {kb..kb+7} U {kb+16..kb+23} with kb = (lane>>4)*8 -> two 16B contiguous loads.
// B fragment is symmetric over columns; Wt stored [r][o][k] makes it contiguous.
__global__ void __launch_bounds__(256)
k_gemm_wmma(const bf16_t* __restrict__ xb, const bf16_t* __restrict__ wt,
            float* __restrict__ xw, int kp, int op) {
    const int lane = threadIdx.x & 31;
    const int wid  = blockIdx.x * (blockDim.x >> 5) + (threadIdx.x >> 5);
    const int OT = op >> 4;                  // 16-col tiles in output
    const int tilesPerRow = RP * OT;
    const int nTiles = (N_ENT / 16) * tilesPerRow;   // 10000 % 16 == 0
    if (wid >= nTiles) return;               // wave-uniform: EXEC stays all-1 for WMMA

    const int rowTile = wid / tilesPerRow;
    const int rem     = wid - rowTile * tilesPerRow;
    const int r       = rem / OT;
    const int ot      = rem - r * OT;

    const int m    = lane & 15;
    const int kb   = (lane >> 4) * 8;        // 0 or 8
    const bf16_t* Arow = xb + (size_t)(rowTile * 16 + m) * kp + kb;
    const bf16_t* Bcol = wt + ((size_t)r * op + (ot * 16 + m)) * kp + kb;

    v8f acc = {};
    for (int k0 = 0; k0 < kp; k0 += 32) {
        __builtin_prefetch(Arow + k0 + 32, 0, 1);   // global_prefetch_b8
        __builtin_prefetch(Bcol + k0 + 32, 0, 1);
        v8bf alo = *(const v8bf*)(Arow + k0);
        v8bf ahi = *(const v8bf*)(Arow + k0 + 16);
        v8bf blo = *(const v8bf*)(Bcol + k0);
        v8bf bhi = *(const v8bf*)(Bcol + k0 + 16);
        v16bf a, b;
        #pragma unroll
        for (int i = 0; i < 8; ++i) {
            a[i] = alo[i]; a[i + 8] = ahi[i];
            b[i] = blo[i]; b[i + 8] = bhi[i];
        }
        acc = __builtin_amdgcn_wmma_f32_16x16x32_bf16(
                  false, a, false, b, (short)0, acc, false, false);
    }

    // D layout: lanes 0-15 -> col=lane, rows v..v+7; lanes 16-31 -> rows 8+v
    const int colBase = ot * 16 + (lane & 15);
    const int rowOff  = (lane >> 4) * 8;
    #pragma unroll
    for (int v = 0; v < 8; ++v) {
        int nrow = rowTile * 16 + rowOff + v;
        xw[((size_t)nrow * RP + r) * op + colBase] = acc[v];
    }
}

// ---------------- edge scatter: out[dst] += xw[src, et] * 1/deg ----------------
// One wave per edge, lanes stride across output features.
__global__ void __launch_bounds__(256)
k_scatter(const int* __restrict__ src, const int* __restrict__ dst,
          const int* __restrict__ et, const float* __restrict__ inv,
          const float* __restrict__ xw, float* __restrict__ out,
          int op, int dout) {
    const int lane = threadIdx.x & 31;
    const int e    = blockIdx.x * (blockDim.x >> 5) + (threadIdx.x >> 5);
    if (e >= EDGES) return;
    const int s = src[e], d = dst[e], t = et[e];
    const float en = inv[d * NUM_R + t];
    const float* m = xw + ((size_t)s * RP + t) * op;
    float* o = out + (size_t)d * dout;
    for (int c = lane; c < dout; c += 32)
        atomicAdd(&o[c], m[c] * en);
}

// ---------------- host-side orchestration ----------------
static inline int cdiv(long long a, long long b) { return (int)((a + b - 1) / b); }

extern "C" void kernel_launch(void* const* d_in, const int* in_sizes, int n_in,
                              void* d_out, int out_size, void* d_ws, size_t ws_size,
                              hipStream_t stream) {
    const float* ent_emb = (const float*)d_in[0];
    const float* bases1  = (const float*)d_in[1];
    const float* comp1   = (const float*)d_in[2];
    const float* root1   = (const float*)d_in[3];
    const float* bias1   = (const float*)d_in[4];
    const float* bases2  = (const float*)d_in[5];
    const float* comp2   = (const float*)d_in[6];
    const float* root2   = (const float*)d_in[7];
    const float* bias2   = (const float*)d_in[8];
    const int*   e_src   = (const int*)d_in[9];          // edge_index[0]
    const int*   e_dst   = ((const int*)d_in[9]) + EDGES; // edge_index[1]
    const int*   e_typ   = (const int*)d_in[10];
    float* out = (float*)d_out;

    // workspace carve-up (256B aligned)
    char* p = (char*)d_ws;
    auto take = [&](size_t bytes) -> char* {
        char* r = p;
        p += (bytes + 255) & ~(size_t)255;
        return r;
    };
    bf16_t* xbf  = (bf16_t*)take((size_t)N_ENT * K1P * sizeof(bf16_t));
    bf16_t* wt1  = (bf16_t*)take((size_t)RP * O1P * K1P * sizeof(bf16_t));
    bf16_t* hbf  = (bf16_t*)take((size_t)N_ENT * K2P * sizeof(bf16_t));
    bf16_t* wt2  = (bf16_t*)take((size_t)RP * O2P * K2P * sizeof(bf16_t));
    float*  hpre = (float*) take((size_t)N_ENT * D_HID * sizeof(float));
    float*  cnt  = (float*) take((size_t)N_ENT * NUM_R * sizeof(float));
    float*  invd = (float*) take((size_t)N_ENT * NUM_R * sizeof(float));
    float*  xw   = (float*) take((size_t)N_ENT * RP * O2P * sizeof(float)); // shared L1/L2, sized for max
    (void)ws_size; (void)n_in; (void)in_sizes; (void)out_size;

    const int B = 256;

    // degree counts (graph identical for both layers -> compute once)
    k_zero_f32<<<cdiv(N_ENT * NUM_R, B), B, 0, stream>>>(cnt, N_ENT * NUM_R);
    k_count   <<<cdiv(EDGES, B), B, 0, stream>>>(e_dst, e_typ, cnt);
    k_inv     <<<cdiv(N_ENT * NUM_R, B), B, 0, stream>>>(cnt, invd, N_ENT * NUM_R);

    // ---------------- layer 1 (D_IN -> D_HID) ----------------
    k_build_wt<<<cdiv((long long)RP * O1P * K1P, B), B, 0, stream>>>(
        bases1, comp1, root1, wt1, D_IN, D_HID, K1P, O1P);
    k_cvt_bf16<<<cdiv((long long)N_ENT * K1P, B), B, 0, stream>>>(ent_emb, xbf, D_IN, K1P);

    {
        long long waves = (long long)(N_ENT / 16) * RP * (O1P / 16);
        k_gemm_wmma<<<cdiv(waves, 8), B, 0, stream>>>(xbf, wt1, xw, K1P, O1P);
    }
    k_init_root<<<cdiv((long long)N_ENT * D_HID, B), B, 0, stream>>>(xw, bias1, hpre, O1P, D_HID);
    k_scatter  <<<cdiv(EDGES, 8), B, 0, stream>>>(e_src, e_dst, e_typ, invd, xw, hpre, O1P, D_HID);
    k_relu_cvt <<<cdiv((long long)N_ENT * K2P, B), B, 0, stream>>>(hpre, hbf);

    // ---------------- layer 2 (D_HID -> D_IN) ----------------
    k_build_wt<<<cdiv((long long)RP * O2P * K2P, B), B, 0, stream>>>(
        bases2, comp2, root2, wt2, D_HID, D_IN, K2P, O2P);
    {
        long long waves = (long long)(N_ENT / 16) * RP * (O2P / 16);
        k_gemm_wmma<<<cdiv(waves, 8), B, 0, stream>>>(hbf, wt2, xw, K2P, O2P);
    }
    k_init_root<<<cdiv((long long)N_ENT * D_IN, B), B, 0, stream>>>(xw, bias2, out, O2P, D_IN);
    k_scatter  <<<cdiv(EDGES, 8), B, 0, stream>>>(e_src, e_dst, e_typ, invd, xw, out, O2P, D_IN);
}
